// wRNN_8203387535481
// MI455X (gfx1250) — matrix operand order repaired and checked by reading, not compiled, not run on hardware
//
#include <hip/hip_runtime.h>
#include <math.h>

// ---------------- problem constants ----------------
#define Vv 4096
#define Ee 1024
#define Mm 4096
#define Ll 2
#define Bb 32
#define Tt 256
#define EPSf 1.1920929e-07f

#define NBLOCKS 64
#define NTHREADS 256

typedef __bf16 bf16;
typedef __attribute__((ext_vector_type(8)))  bf16  v8bf;
typedef __attribute__((ext_vector_type(16))) bf16  v16bf;
typedef __attribute__((ext_vector_type(8)))  float v8f;

// ---------------- workspace layout (bytes) ----------------
#define WS_CTRL  ((size_t)0)                       // 2 x u32 barrier state (256B reserved)
#define WS_X     ((size_t)256)                     // f32 [32,1024]  = 128 KB
#define WS_H     (WS_X  + (size_t)Bb*Ee*4)         // f32 [32,1024]  = 128 KB
#define WS_HN    (WS_H  + (size_t)Bb*Ee*4)         // bf16 [32,1024] = 64 KB
#define WS_U     (WS_HN + (size_t)Bb*Ee*2)         // bf16 [32,4096] = 256 KB
#define WS_XN    (WS_U  + (size_t)Bb*Mm*2)         // bf16 [32,1024] = 64 KB
#define WS_EMB   (WS_XN + (size_t)Bb*Ee*2)         // bf16 [4096,1024] = 8 MB
#define WS_FC1   (WS_EMB + (size_t)Vv*Ee*2)        // bf16 [2,4096,1024] = 16 MB
#define WS_FC2   (WS_FC1 + (size_t)Ll*Mm*Ee*2)     // bf16 [2,1024,4096] = 16 MB

// ---------------- grid barrier (generation counter = sense) ----------------
__device__ __forceinline__ void grid_barrier(unsigned* cnt, unsigned* gen,
                                             unsigned epoch) {
  __syncthreads();
  if (threadIdx.x == 0) {
    __threadfence();
    if (atomicAdd(cnt, 1u) == (unsigned)(gridDim.x - 1)) {
      *cnt = 0u;
      __threadfence();
      atomicAdd(gen, 1u);
    } else {
      while (atomicAdd(gen, 0u) <= epoch) __builtin_amdgcn_s_sleep(1);
    }
  }
  __syncthreads();
}

// ---------------- 16x16 bf16 WMMA tile over K ----------------
// A 16xK (row-major, activations): lane&15 = row M, lane>>4 selects K-half:
//   lanes 0-15 hold K {0..7, 16..23}, lanes 16-31 hold K {8..15, 24..31}
//   (per ISA 16-bit A table) -> two 16B loads at +ks and +ks+16.
// B Kx16 (weight rows = output cols, row-major): lane&15 = col N, lane>>4
//   selects K {0..7, 16..23} / {8..15, 24..31} (same as A) ->
//   two 16B loads at +ks and +ks+16.
__device__ __forceinline__ void gemm_tile_bf16(const bf16* __restrict__ Abase, int lda,
                                               const bf16* __restrict__ Bbase, int ldb,
                                               int K, int lane, v8f& acc) {
  const int rc  = lane & 15;
  const int ksA = (lane >> 4) << 3;   // 0 or 8
  const int ksB = (lane >> 4) << 3;   // 0 or 8
  const bf16* pa = Abase + (size_t)rc * lda + ksA;
  const bf16* pb = Bbase + (size_t)rc * ldb + ksB;
#pragma unroll 4
  for (int k = 0; k < K; k += 32) {
    __builtin_prefetch(pb + 512, 0, 1);  // global_prefetch_b8 on weight stream
    v8bf alo = *(const v8bf*)pa;
    v8bf ahi = *(const v8bf*)(pa + 16);
    v8bf blo = *(const v8bf*)pb;
    v8bf bhi = *(const v8bf*)(pb + 16);
    v16bf b = __builtin_shufflevector(blo, bhi, 0,1,2,3,4,5,6,7,8,9,10,11,12,13,14,15);
    v16bf a = __builtin_shufflevector(alo, ahi, 0,1,2,3,4,5,6,7,8,9,10,11,12,13,14,15);
    acc = __builtin_amdgcn_wmma_f32_16x16x32_bf16(false, a, false, b,
                                                  (short)0, acc, false, false);
    pa += 32;
    pb += 32;
  }
}

// RMSNorm of one row (length Ee) by one wave32; optionally write f32 in place
template <bool WRITE_F32>
__device__ __forceinline__ void rmsnorm_row(const float* __restrict__ src,
                                            const float* __restrict__ w,
                                            float* dst_f32,
                                            bf16* __restrict__ dst_bf, int lane) {
  float s = 0.f;
  for (int e = lane; e < Ee; e += 32) { float v = src[e]; s += v * v; }
#pragma unroll
  for (int o = 16; o; o >>= 1) s += __shfl_xor(s, o, 32);
  const float scale = rsqrtf(s * (1.0f / Ee) + EPSf);
  for (int e = lane; e < Ee; e += 32) {
    float v = src[e] * scale * w[e];
    if (WRITE_F32) dst_f32[e] = v;
    dst_bf[e] = (bf16)v;
  }
}

// ---------------- kernels ----------------
__global__ void init_ctrl_kernel(unsigned* ctrl, float* loss_slot) {
  if (threadIdx.x == 0) { ctrl[0] = 0u; ctrl[1] = 0u; *loss_slot = 0.0f; }
}

__global__ __launch_bounds__(NTHREADS) void convert_weights_kernel(
    const float* __restrict__ emb, const float* __restrict__ fc1,
    const float* __restrict__ fc2, bf16* __restrict__ embB,
    bf16* __restrict__ fc1B, bf16* __restrict__ fc2B) {
  const int tid = blockIdx.x * blockDim.x + threadIdx.x;
  const int n = gridDim.x * blockDim.x;
  for (int j = tid; j < Vv * Ee; j += n) embB[j] = (bf16)emb[j];
  for (int j = tid; j < Ll * Mm * Ee; j += n) fc1B[j] = (bf16)fc1[j];
  for (int j = tid; j < Ll * Ee * Mm; j += n) fc2B[j] = (bf16)fc2[j];
}

__global__ __launch_bounds__(NTHREADS) void rnn_scan_kernel(
    const int* __restrict__ tokens, const float* __restrict__ emb_w,
    const float* __restrict__ norm_w, const float* __restrict__ fc1_b,
    const float* __restrict__ fc2_b, const float* __restrict__ final_norm_w,
    char* __restrict__ ws, float* __restrict__ logits_out) {
  unsigned* cnt = (unsigned*)(ws + WS_CTRL);
  unsigned* gen = cnt + 1;
  float* x = (float*)(ws + WS_X);
  float* h = (float*)(ws + WS_H);
  bf16* hn = (bf16*)(ws + WS_HN);
  bf16* u = (bf16*)(ws + WS_U);
  bf16* xn = (bf16*)(ws + WS_XN);
  const bf16* embB = (const bf16*)(ws + WS_EMB);
  const bf16* fc1B = (const bf16*)(ws + WS_FC1);
  const bf16* fc2B = (const bf16*)(ws + WS_FC2);

  // split-K(4) partial-tile exchange for fc2 (deterministic in-block reduce)
  __shared__ float redD[2][4][256];

  const int tid = blockIdx.x * blockDim.x + threadIdx.x;
  const int nthr = gridDim.x * blockDim.x;
  const int wid = tid >> 5;
  const int nwaves = nthr >> 5;
  const int lane = threadIdx.x & 31;
  const int wv = threadIdx.x >> 5;  // block-local wave 0..7
  unsigned epoch = 0;

  // x = 0
  for (int i = tid; i < Bb * Ee; i += nthr) x[i] = 0.0f;
  grid_barrier(cnt, gen, epoch++);

  for (int t = 0; t < Tt; ++t) {
    // ---- Phase A: x += emb[tok]; h = x
    for (int i = tid; i < Bb * Ee; i += nthr) {
      int b = i >> 10, e = i & (Ee - 1);
      int tok = tokens[b * Tt + t];
      float v = x[i] + emb_w[(size_t)tok * Ee + e];
      x[i] = v;
      h[i] = v;
    }
    grid_barrier(cnt, gen, epoch++);

    for (int l = 0; l < Ll; ++l) {
      // ---- Phase B: hn = rmsnorm(h, norm_w[l]) -> bf16
      for (int row = wid; row < Bb; row += nwaves)
        rmsnorm_row<false>(h + (size_t)row * Ee, norm_w + (size_t)l * Ee,
                           nullptr, hn + (size_t)row * Ee, lane);
      grid_barrier(cnt, gen, epoch++);

      // ---- Phase C: u = silu(hn @ fc1^T + b1) -> bf16  (tiles: 2 x 256)
      for (int tile = wid; tile < (Bb / 16) * (Mm / 16); tile += nwaves) {
        int rt = tile >> 8, ct = tile & 255;
        v8f acc = {};
        gemm_tile_bf16(hn + (size_t)rt * 16 * Ee, Ee,
                       fc1B + (size_t)l * Mm * Ee + (size_t)ct * 16 * Ee, Ee,
                       Ee, lane, acc);
        int col = lane & 15, rowb = (lane >> 4) << 3;
        int m = ct * 16 + col;
        float bias = fc1_b[l * Mm + m];
#pragma unroll
        for (int rr = 0; rr < 8; ++rr) {
          int b = rt * 16 + rowb + rr;
          float v = acc[rr] + bias;
          float sv = v * __builtin_amdgcn_rcpf(1.0f + __expf(-v));  // fast SiLU
          u[(size_t)b * Mm + m] = (bf16)sv;
        }
      }
      grid_barrier(cnt, gen, epoch++);

      // ---- Phase D: h += u @ fc2^T + b2 (128 tiles, K=4096 split x4 in-block;
      //      last layer also fuses outer residual x += h)
      for (int tp = blockIdx.x; tp < ((Bb / 16) * (Ee / 16)) / 2; tp += gridDim.x) {
        const int tile = tp * 2 + (wv >> 2);  // 2 tiles per block
        const int kc = wv & 3;                // K chunk 0..3 (1024 each)
        const int rt = tile >> 6, ct = tile & 63;
        v8f acc = {};
        gemm_tile_bf16(u + (size_t)rt * 16 * Mm + (size_t)kc * 1024, Mm,
                       fc2B + (size_t)l * Ee * Mm + (size_t)ct * 16 * Mm +
                           (size_t)kc * 1024,
                       Mm, 1024, lane, acc);
        int col = lane & 15, rowb = (lane >> 4) << 3;
#pragma unroll
        for (int rr = 0; rr < 8; ++rr)
          redD[wv >> 2][kc][(rowb + rr) * 16 + col] = acc[rr];
        __syncthreads();
        // deterministic reduce of the 4 K-partials; 512 elems / 256 threads
        for (int i = threadIdx.x; i < 512; i += NTHREADS) {
          int tl = i >> 8, el = i & 255;
          int gtile = tp * 2 + tl;
          int grt = gtile >> 6, gct = gtile & 63;
          int row = el >> 4, cc = el & 15;
          int b = grt * 16 + row, e = gct * 16 + cc;
          float s = redD[tl][0][el] + redD[tl][1][el] + redD[tl][2][el] +
                    redD[tl][3][el] + fc2_b[l * Ee + e];
          size_t idx = (size_t)b * Ee + e;
          float hf = h[idx] + s;
          h[idx] = hf;
          if (l == Ll - 1) x[idx] += hf;  // fused outer residual
        }
        __syncthreads();
      }
      grid_barrier(cnt, gen, epoch++);
    }

    // ---- Phase E: x = rmsnorm(x, final_norm_w) in place; xn = bf16(x)
    for (int row = wid; row < Bb; row += nwaves)
      rmsnorm_row<true>(x + (size_t)row * Ee, final_norm_w,
                        x + (size_t)row * Ee, xn + (size_t)row * Ee, lane);
    grid_barrier(cnt, gen, epoch++);

    // ---- Phase F: logits[:, t, :] = xn @ emb^T  (tiles: 2 x 256)
    // no trailing barrier: next Phase A only touches x/h, and its barrier
    // orders Phase F before any later reader/writer of xn / d_out.
    for (int tile = wid; tile < (Bb / 16) * (Vv / 16); tile += nwaves) {
      int rt = tile >> 8, ct = tile & 255;
      v8f acc = {};
      gemm_tile_bf16(xn + (size_t)rt * 16 * Ee, Ee,
                     embB + (size_t)ct * 16 * Ee, Ee, Ee, lane, acc);
      int col = lane & 15, rowb = (lane >> 4) << 3;
      int vc = ct * 16 + col;
#pragma unroll
      for (int rr = 0; rr < 8; ++rr) {
        int b = rt * 16 + rowb + rr;
        logits_out[((size_t)b * Tt + t) * Vv + vc] = acc[rr];
      }
    }
  }
}

__global__ __launch_bounds__(NTHREADS) void ce_loss_kernel(
    const float* __restrict__ logits, const int* __restrict__ targets,
    float* __restrict__ loss_out) {
  const int row = blockIdx.x;  // b*T + t
  const float* lrow = logits + (size_t)row * Vv;
  const int lane = threadIdx.x & 31;
  const int warp = threadIdx.x >> 5;
  const int nwarp = blockDim.x >> 5;
  __shared__ float red[8];

  float m = -3.4e38f;
  for (int v = threadIdx.x; v < Vv; v += blockDim.x) m = fmaxf(m, lrow[v]);
#pragma unroll
  for (int o = 16; o; o >>= 1) m = fmaxf(m, __shfl_xor(m, o, 32));
  if (lane == 0) red[warp] = m;
  __syncthreads();
  float bm = red[0];
  for (int i = 1; i < nwarp; ++i) bm = fmaxf(bm, red[i]);
  __syncthreads();

  float s = 0.f;
  for (int v = threadIdx.x; v < Vv; v += blockDim.x) s += __expf(lrow[v] - bm);
#pragma unroll
  for (int o = 16; o; o >>= 1) s += __shfl_xor(s, o, 32);
  if (lane == 0) red[warp] = s;
  __syncthreads();
  if (threadIdx.x == 0) {
    float ss = 0.f;
    for (int i = 0; i < nwarp; ++i) ss += red[i];
    float lse = bm + __logf(ss);
    float li = lrow[targets[row]];
    atomicAdd(loss_out, (lse - li) * (1.0f / (Bb * Tt)));
  }
}

// ---------------- launcher ----------------
extern "C" void kernel_launch(void* const* d_in, const int* in_sizes, int n_in,
                              void* d_out, int out_size, void* d_ws, size_t ws_size,
                              hipStream_t stream) {
  const int* tokens = (const int*)d_in[0];
  const int* targets = (const int*)d_in[1];
  const float* emb_w = (const float*)d_in[2];
  const float* norm_w = (const float*)d_in[3];
  const float* fc1_w = (const float*)d_in[4];
  const float* fc1_b = (const float*)d_in[5];
  const float* fc2_w = (const float*)d_in[6];
  const float* fc2_b = (const float*)d_in[7];
  const float* fnw = (const float*)d_in[8];

  float* logits = (float*)d_out;
  float* loss_slot = logits + (size_t)Bb * Tt * Vv;
  char* ws = (char*)d_ws;

  unsigned* ctrl = (unsigned*)(ws + WS_CTRL);
  bf16* embB = (bf16*)(ws + WS_EMB);
  bf16* fc1B = (bf16*)(ws + WS_FC1);
  bf16* fc2B = (bf16*)(ws + WS_FC2);

  // persistent-kernel residency: never launch more blocks than compute units
  int dev = 0;
  (void)hipGetDevice(&dev);
  int ncu = NBLOCKS;
  (void)hipDeviceGetAttribute(&ncu, hipDeviceAttributeMultiprocessorCount, dev);
  int nb = (ncu < NBLOCKS) ? ncu : NBLOCKS;
  if (nb < 1) nb = 1;

  init_ctrl_kernel<<<1, 32, 0, stream>>>(ctrl, loss_slot);
  convert_weights_kernel<<<2048, NTHREADS, 0, stream>>>(emb_w, fc1_w, fc2_w,
                                                        embB, fc1B, fc2B);
  rnn_scan_kernel<<<nb, NTHREADS, 0, stream>>>(tokens, emb_w, norm_w, fc1_b,
                                               fc2_b, fnw, ws, logits);
  ce_loss_kernel<<<Bb * Tt, NTHREADS, 0, stream>>>(logits, targets, loss_slot);
}
